// MultiHeadedAttention_20675972563719
// MI455X (gfx1250) — matrix operand order repaired and checked by reading, not compiled, not run on hardware
//
#include <hip/hip_runtime.h>

typedef __bf16 bf16;
typedef __attribute__((ext_vector_type(16))) __bf16 v16bf;
typedef __attribute__((ext_vector_type(8)))  float  v8f;

constexpr int Bc = 4, Sc = 1024, Dc = 1024, Hc = 16, DKc = 64;
constexpr int Mtot = Bc * Sc; // 4096

// ---------------- WMMA fragment loaders (CDNA5 16x16x32 bf16 layouts) ----------------
// A (16x32, MxK), LDS row-major [m][k] with row stride ldk:
//   lane m=lane&15, half=lane>>4; elems 0..7 = A[m][8*half + 0..7], elems 8..15 = A[m][16+8*half+0..7]
__device__ __forceinline__ v16bf load_a_frag(const bf16* p0, int ldk, int lane) {
    const int half = lane >> 4, m = lane & 15;
    const bf16* r = p0 + m * ldk;
    union { v16bf v; uint4 q[2]; } u;
    u.q[0] = *(const uint4*)(r + half * 8);
    u.q[1] = *(const uint4*)(r + 16 + half * 8);
    return u.v;
}
// B (32x16, KxN), LDS stored K-contiguous per column: [n][k] with row stride ldk:
//   lane n=lane&15, half=lane>>4; elems 0..7 = lds[n][8*half + 0..7], elems 8..15 = lds[n][16+8*half+0..7]
__device__ __forceinline__ v16bf load_b_frag(const bf16* p0, int ldk, int lane) {
    const int half = lane >> 4, n = lane & 15;
    const bf16* r = p0 + n * ldk;
    union { v16bf v; uint4 q[2]; } u;
    u.q[0] = *(const uint4*)(r + half * 8);
    u.q[1] = *(const uint4*)(r + 16 + half * 8);
    return u.v;
}

#define WMMA_BF16(a, b, c) \
    __builtin_amdgcn_wmma_f32_16x16x32_bf16(false, (a), false, (b), (short)0, (c), false, false)

// ---- CDNA5 async global->LDS copy: 64 bytes per lane (4 x b128), ASYNCcnt-tracked.
// ISA 15.18.3: INST_OFFSET is added to BOTH the LDS (VDST) and global (VADDR) address,
// so one base pair covers the whole 64B chunk via offset:0/16/32/48.
__device__ __forceinline__ void async_copy_64B(unsigned lds_off, const void* gptr) {
    asm volatile(
        "global_load_async_to_lds_b128 %0, %1, off\n\t"
        "global_load_async_to_lds_b128 %0, %1, off offset:16\n\t"
        "global_load_async_to_lds_b128 %0, %1, off offset:32\n\t"
        "global_load_async_to_lds_b128 %0, %1, off offset:48"
        :: "v"(lds_off), "v"(gptr) : "memory");
}
__device__ __forceinline__ void wait_async0() {
    asm volatile("s_wait_asynccnt 0x0" ::: "memory");
}
// Generic (flat) LDS address -> wave-relative LDS byte offset: low 32 bits (ISA 10.2 aperture map)
__device__ __forceinline__ unsigned lds_off_of(const void* p) {
    return (unsigned)(unsigned long long)p;
}

// ---------------- Kernel 1: C[M,N] = A[M,K] @ W[N,K]^T + bias, bf16 out ----------------
// Block: 256 thr (8 waves), tile 128x128, waves as 4(M) x 2(N), each wave 32x64.
__global__ __launch_bounds__(256) void gemm_xWT_bias_bf16(
    const float* __restrict__ A, const float* __restrict__ W,
    const float* __restrict__ bias, bf16* __restrict__ out,
    int M, int N, int K)
{
    __shared__ __align__(16) bf16 lA[128][32]; // 8 KB  [m][k]
    __shared__ __align__(16) bf16 lB[128][32]; // 8 KB  [n][k]
    const int t = threadIdx.x;
    const int lane = t & 31, wave = t >> 5;
    const int wm = wave >> 1, wn = wave & 1;
    const int m0 = blockIdx.y * 128, n0 = blockIdx.x * 128;

    v8f acc[2][4];
#pragma unroll
    for (int i = 0; i < 2; i++)
#pragma unroll
        for (int j = 0; j < 4; j++) acc[i][j] = (v8f){0,0,0,0,0,0,0,0};

    const int sr = t >> 1, sc = (t & 1) * 16;
    // hoisted staging pointers; advance by 32 floats per K-step (no per-iter 64-bit re-derivation)
    const float* ga = A + (size_t)(m0 + sr) * K + sc;
    const float* gw = W + (size_t)(n0 + sr) * K + sc;

    for (int k0 = 0; k0 < K; k0 += 32, ga += 32, gw += 32) {
        __syncthreads();
        { // stage f32 -> bf16 into LDS (A natural, W natural [n][k] == B^T layout)
            float fa[16], fw[16];
#pragma unroll
            for (int i = 0; i < 4; i++) {
                ((float4*)fa)[i] = ((const float4*)ga)[i];
                ((float4*)fw)[i] = ((const float4*)gw)[i];
            }
            // prefetch next K tile into cache while this step's WMMAs run
            __builtin_prefetch(ga + 32, 0, 1);
            __builtin_prefetch(gw + 32, 0, 1);
#pragma unroll
            for (int i = 0; i < 16; i++) {
                lA[sr][sc + i] = (bf16)fa[i];
                lB[sr][sc + i] = (bf16)fw[i];
            }
        }
        __syncthreads();
        v16bf af[2], bfv[4];
#pragma unroll
        for (int i = 0; i < 2; i++) af[i] = load_a_frag(&lA[wm * 32 + i * 16][0], 32, lane);
#pragma unroll
        for (int j = 0; j < 4; j++) bfv[j] = load_b_frag(&lB[wn * 64 + j * 16][0], 32, lane);
#pragma unroll
        for (int i = 0; i < 2; i++)
#pragma unroll
            for (int j = 0; j < 4; j++)
                acc[i][j] = WMMA_BF16(af[i], bfv[j], acc[i][j]);
    }

    const int half = lane >> 4, nl = lane & 15;
#pragma unroll
    for (int j = 0; j < 4; j++) {
        const int n = n0 + wn * 64 + j * 16 + nl;
        const float bv = bias[n];
#pragma unroll
        for (int i = 0; i < 2; i++)
#pragma unroll
            for (int r = 0; r < 8; r++) {
                const int m = m0 + wm * 32 + i * 16 + r + 8 * half;
                out[(size_t)m * N + n] = (bf16)(acc[i][j][r] + bv);
            }
    }
}

// ---------------- Kernel 2: flash attention, one (b,h,128-q block) per block ----------------
__global__ __launch_bounds__(256) void attn_fwd(
    const bf16* __restrict__ Qp, const bf16* __restrict__ Kp, const bf16* __restrict__ Vp,
    const int* __restrict__ mask, float* __restrict__ X)
{
    __shared__ __align__(16) bf16 lK[128][64];  // 16 KB [key][d]  (== B-frag layout for Q K^T)
    __shared__ __align__(16) bf16 lVt[64][128]; // 16 KB [d][key]  (== B-frag layout for P V)
    __shared__ __align__(16) bf16 lP[128][128]; // 32 KB probs; front also used to stage Q

    const int t = threadIdx.x, lane = t & 31, w = t >> 5;
    const int half = lane >> 4, nl = lane & 15;

    const int bid = blockIdx.x;
    const int qb = bid & 7, bh = bid >> 3;
    const int b = bh >> 4, h = bh & 15;
    const int q0 = qb * 128;

    const size_t headOff = ((size_t)b * Sc) * Dc + (size_t)h * DKc;
    const bf16* Qh = Qp + headOff;
    const bf16* Kh = Kp + headOff;
    const bf16* Vh = Vp + headOff;

    const int row = t >> 1, dbase = (t & 1) * 32; // staging: 64B per thread per tile row

    // ---- stage Q block [128][64] into front of lP via async copy, pull wave's A-frags ----
    bf16* lQ = &lP[0][0];
    async_copy_64B(lds_off_of(lQ + row * 64 + dbase),
                   Qh + (size_t)(q0 + row) * Dc + dbase);
    wait_async0();
    __syncthreads();
    v16bf a_q[2];
#pragma unroll
    for (int ks = 0; ks < 2; ks++) a_q[ks] = load_a_frag(lQ + (w * 16) * 64 + ks * 32, 64, lane);
    __syncthreads();

    float rmax[8], rsum[8];
    v8f o[4];
#pragma unroll
    for (int r = 0; r < 8; r++) { rmax[r] = -3.0e38f; rsum[r] = 0.f; }
#pragma unroll
    for (int nt = 0; nt < 4; nt++) o[nt] = (v8f){0,0,0,0,0,0,0,0};

    // hoisted staging pointers, advanced by one key-block of rows per iteration
    const bf16* kg = Kh + (size_t)row * Dc + dbase;
    const bf16* vg = Vh + (size_t)row * Dc + dbase;
    const unsigned lKoff = lds_off_of(&lK[row][dbase]);
    constexpr size_t kbStride = (size_t)128 * Dc;

    for (int kb = 0; kb < Sc / 128; kb++, kg += kbStride, vg += kbStride) {
        __syncthreads();
        // K tile: pure copy -> async global->LDS (ASYNCcnt path, no VGPR round-trip)
        async_copy_64B(lKoff, kg);
        // V tile needs transpose: through VGPRs, scatter to lVt[d][key]
        {
            bf16 vv[32];
#pragma unroll
            for (int i = 0; i < 4; i++) ((uint4*)vv)[i] = ((const uint4*)vg)[i];
            __builtin_prefetch(kg + kbStride, 0, 1);
            __builtin_prefetch(vg + kbStride, 0, 1);
#pragma unroll
            for (int i = 0; i < 32; i++) lVt[dbase + i][row] = vv[i];
        }
        wait_async0();
        __syncthreads();

        // scores: wave's 16 q-rows vs 128 keys
        v8f s[8];
#pragma unroll
        for (int ct = 0; ct < 8; ct++) {
            v8f acc = (v8f){0,0,0,0,0,0,0,0};
#pragma unroll
            for (int ks = 0; ks < 2; ks++) {
                v16bf bk = load_b_frag(&lK[ct * 16][ks * 32], 64, lane);
                acc = WMMA_BF16(a_q[ks], bk, acc);
            }
            s[ct] = acc;
        }

        // scale + mask + online softmax (row = r + 8*half, col = lane&15 within tile)
        const int qrow_base = q0 + w * 16;
#pragma unroll
        for (int r = 0; r < 8; r++) {
            const int q = qrow_base + r + 8 * half;
            const size_t mbase = ((size_t)b * Sc + q) * Sc + (size_t)kb * 128 + nl;
            float mx = -3.0e38f;
#pragma unroll
            for (int ct = 0; ct < 8; ct++) {
                float v = s[ct][r] * 0.125f; // 1/sqrt(64)
                if (mask[mbase + ct * 16] == 0) v = -1.0e9f;
                s[ct][r] = v;
                mx = fmaxf(mx, v);
            }
#pragma unroll
            for (int off = 1; off < 16; off <<= 1) mx = fmaxf(mx, __shfl_xor(mx, off, 32));
            const float mnew = fmaxf(rmax[r], mx);
            const float corr = __expf(rmax[r] - mnew);
            rmax[r] = mnew;
            float ls = 0.f;
            const int prow = w * 16 + r + 8 * half;
#pragma unroll
            for (int ct = 0; ct < 8; ct++) {
                float p = __expf(s[ct][r] - mnew);
                ls += p;
                lP[prow][ct * 16 + nl] = (bf16)p;
            }
#pragma unroll
            for (int off = 1; off < 16; off <<= 1) ls += __shfl_xor(ls, off, 32);
            rsum[r] = rsum[r] * corr + ls;
#pragma unroll
            for (int nt = 0; nt < 4; nt++) o[nt][r] *= corr;
        }

        // O += P @ V   (lP rows are wave-private; per-wave DS ordering suffices)
#pragma unroll
        for (int ks = 0; ks < 4; ks++) {
            v16bf ap = load_a_frag(&lP[w * 16][ks * 32], 128, lane);
#pragma unroll
            for (int nt = 0; nt < 4; nt++) {
                v16bf bv = load_b_frag(&lVt[nt * 16][ks * 32], 128, lane);
                o[nt] = WMMA_BF16(ap, bv, o[nt]);
            }
        }
    }

    // normalize + write X[b, q, h*64 + d] in f32
#pragma unroll
    for (int nt = 0; nt < 4; nt++)
#pragma unroll
        for (int r = 0; r < 8; r++) {
            const int q = q0 + w * 16 + r + 8 * half;
            const int d = h * DKc + nt * 16 + nl;
            X[((size_t)b * Sc + q) * Dc + d] = o[nt][r] / rsum[r];
        }
}

// ---------------- Kernel 3: w̄ = mean_rows(Wo), b̄ = mean(bo) ----------------
__global__ __launch_bounds__(256) void reduce_wo(const float* __restrict__ Wo,
                                                 const float* __restrict__ bo,
                                                 float* __restrict__ wbar,
                                                 float* __restrict__ bbar)
{
    const int k = blockIdx.x * 256 + threadIdx.x;
    float s = 0.f;
    for (int j = 0; j < Dc; j++) s += Wo[(size_t)j * Dc + k];
    wbar[k] = s * (1.0f / Dc);
    if (k == 0) {
        float tb = 0.f;
        for (int j = 0; j < Dc; j++) tb += bo[j];
        *bbar = tb * (1.0f / Dc);
    }
}

// ---------------- Kernel 4: t = thresh(x); out = t + (t·w̄ + b̄) ----------------
__global__ __launch_bounds__(256) void finalize_k(const float* __restrict__ X,
                                                  const float* __restrict__ wbar,
                                                  const float* __restrict__ bbar,
                                                  float* __restrict__ out)
{
    __shared__ float red[256];
    const int row = blockIdx.x, t = threadIdx.x;
    const float* x = X + (size_t)row * Dc;
    float tv[4];
    float part = 0.f;
#pragma unroll
    for (int i = 0; i < 4; i++) {
        const int c = i * 256 + t;
        float v = x[c];
        v = (v > 0.2f) ? v : 0.f; // F.threshold(x, 0.2, 0.0)
        tv[i] = v;
        part += v * wbar[c];
    }
    red[t] = part;
    __syncthreads();
    for (int sdx = 128; sdx > 0; sdx >>= 1) {
        if (t < sdx) red[t] += red[t + sdx];
        __syncthreads();
    }
    const float mean = red[0] + *bbar;
#pragma unroll
    for (int i = 0; i < 4; i++) out[(size_t)row * Dc + i * 256 + t] = tv[i] + mean;
}

// ---------------- launcher ----------------
extern "C" void kernel_launch(void* const* d_in, const int* in_sizes, int n_in,
                              void* d_out, int out_size, void* d_ws, size_t ws_size,
                              hipStream_t stream) {
    (void)in_sizes; (void)n_in; (void)out_size; (void)ws_size;
    const float* query = (const float*)d_in[0];
    const float* key   = (const float*)d_in[1];
    const float* value = (const float*)d_in[2];
    const int*   mask  = (const int*)d_in[3];
    const float* Wq = (const float*)d_in[4];
    const float* bq = (const float*)d_in[5];
    const float* Wk = (const float*)d_in[6];
    const float* bk = (const float*)d_in[7];
    const float* Wv = (const float*)d_in[8];
    const float* bv = (const float*)d_in[9];
    const float* Wo = (const float*)d_in[10];
    const float* bo = (const float*)d_in[11];

    char* ws = (char*)d_ws;
    const size_t MB = 1024ull * 1024ull;
    bf16*  Qp   = (bf16*)(ws + 0);        //  8 MB
    bf16*  Kp   = (bf16*)(ws + 8 * MB);   //  8 MB
    bf16*  Vp   = (bf16*)(ws + 16 * MB);  //  8 MB
    float* X    = (float*)(ws + 24 * MB); // 16 MB
    float* wbar = (float*)(ws + 40 * MB); //  4 KB
    float* bbar = wbar + Dc;

    dim3 gg(Dc / 128, Mtot / 128); // (8, 32)
    gemm_xWT_bias_bf16<<<gg, 256, 0, stream>>>(query, Wq, bq, Qp, Mtot, Dc, Dc);
    gemm_xWT_bias_bf16<<<gg, 256, 0, stream>>>(key,   Wk, bk, Kp, Mtot, Dc, Dc);
    gemm_xWT_bias_bf16<<<gg, 256, 0, stream>>>(value, Wv, bv, Vp, Mtot, Dc, Dc);

    attn_fwd<<<Bc * Hc * (Sc / 128), 256, 0, stream>>>(Qp, Kp, Vp, mask, X);

    reduce_wo<<<Dc / 256, 256, 0, stream>>>(Wo, bo, wbar, bbar);
    finalize_k<<<Mtot, 256, 0, stream>>>(X, wbar, bbar, (float*)d_out);
}
